// Variant3Block_87153476370488
// MI455X (gfx1250) — compile-verified, matrix operand order checked by reading, not run on hardware
//
#include <hip/hip_runtime.h>

// ---------------------------------------------------------------- types
typedef __attribute__((ext_vector_type(16))) _Float16 v16h;
typedef __attribute__((ext_vector_type(8)))  _Float16 v8h;
typedef __attribute__((ext_vector_type(4)))  _Float16 v4h;
typedef __attribute__((ext_vector_type(8)))  float    v8f;
typedef __attribute__((ext_vector_type(4)))  float    v4f;
typedef __attribute__((ext_vector_type(4)))  int      v4i;

#define D_MODEL 1024
#define T_SEQ   1024
#define NBATCH  4
#define N_TOK   4096   /* B*T */
#define D_FF    4096
#define N_HEADS 16

#define WMMA_F16(a, b, c) \
  __builtin_amdgcn_wmma_f32_16x16x32_f16(false, (a), false, (b), (short)0, (c), false, false)

#if __has_builtin(__builtin_amdgcn_global_load_async_to_lds_b128) && \
    __has_builtin(__builtin_amdgcn_s_wait_asynccnt)
#define USE_ASYNC_LDS 1
typedef __attribute__((address_space(1))) v4i gv4i;
typedef __attribute__((address_space(3))) v4i lv4i;
#else
#define USE_ASYNC_LDS 0
#endif

// ---------------------------------------------------------------- helpers
static __device__ __forceinline__ float sigmf(float x) { return 1.0f / (1.0f + __expf(-x)); }

static __device__ __forceinline__ float redmax16(float v) {
#pragma unroll
  for (int m = 1; m < 16; m <<= 1) v = fmaxf(v, __shfl_xor(v, m, 32));
  return v;
}
static __device__ __forceinline__ float redsum16(float v) {
#pragma unroll
  for (int m = 1; m < 16; m <<= 1) v += __shfl_xor(v, m, 32);
  return v;
}
static __device__ __forceinline__ float redmax32(float v) {
#pragma unroll
  for (int m = 1; m < 32; m <<= 1) v = fmaxf(v, __shfl_xor(v, m, 32));
  return v;
}
static __device__ __forceinline__ float redsum32(float v) {
#pragma unroll
  for (int m = 1; m < 32; m <<= 1) v += __shfl_xor(v, m, 32);
  return v;
}

// A/B fragment (16x32 / 32x16, f16) from LDS tile, padded row stride 40 halves.
// lane<16: row = lane,    k in {0..7, 16..23}
// lane>=16: row = lane-16, k in {8..15, 24..31}
static __device__ __forceinline__ v16h frag_lds40(const _Float16* base, int lane) {
  int l = lane & 15, hi = (lane >> 4) & 1;
  const _Float16* p = base + l * 40 + hi * 8;
  v8h lo  = *(const v8h*)(p);
  v8h hi8 = *(const v8h*)(p + 16);
  v16h r;
#pragma unroll
  for (int i = 0; i < 8; ++i) { r[i] = lo[i]; r[i + 8] = hi8[i]; }
  return r;
}

// Same fragment pattern from row-major f16 global memory (ld = row stride in halves).
static __device__ __forceinline__ v16h frag_g16(const _Float16* p0, int ld, int lane) {
  int l = lane & 15, hi = (lane >> 4) & 1;
  const _Float16* p = p0 + (size_t)l * ld + hi * 8;
  v8h lo  = *(const v8h*)(p);
  v8h hi8 = *(const v8h*)(p + 16);
  v16h r;
#pragma unroll
  for (int i = 0; i < 8; ++i) { r[i] = lo[i]; r[i + 8] = hi8[i]; }
  return r;
}

// B fragment (32x16): column n in lane n, reading a K-major f16 matrix
// (element [k][n] at p0 + k*ld + n). Used for P@V where V rows are the K dim.
static __device__ __forceinline__ v16h frag_vcol16(const _Float16* p0, int ld, int lane) {
  int n = lane & 15, hi = (lane >> 4) & 1;
  const _Float16* p = p0 + n + (size_t)(hi * 8) * ld;
  v16h r;
#pragma unroll
  for (int i = 0; i < 8; ++i) {
    r[i]     = p[(size_t)i * ld];
    r[i + 8] = p[(size_t)(i + 16) * ld];
  }
  return r;
}

// ---------------------------------------------------------------- converters
// f32 -> f16, n must be a multiple of 4; one v4 per thread.
__global__ __launch_bounds__(256) void cvt16_kernel(const float* __restrict__ in,
                                                    _Float16* __restrict__ out) {
  int idx = (blockIdx.x * 256 + threadIdx.x) * 4;
  v4f a = *(const v4f*)(in + idx);
  v4h h;
#pragma unroll
  for (int i = 0; i < 4; ++i) h[i] = (_Float16)a[i];
  *(v4h*)(out + idx) = h;
}

// an_embed (64,1024) f32 -> transposed (1024,64) f16
__global__ __launch_bounds__(256) void tconv_kernel(const float* __restrict__ in,
                                                    _Float16* __restrict__ out) {
  int idx = blockIdx.x * 256 + threadIdx.x;  // over 1024*64
  int n = idx >> 6, k = idx & 63;
  out[idx] = (_Float16)in[k * 1024 + n];
}

// ---------------------------------------------------------------- LayerNorm
__global__ __launch_bounds__(256) void ln_kernel(const float* __restrict__ x,
                                                 const float* __restrict__ g,
                                                 const float* __restrict__ b,
                                                 float* __restrict__ y32,
                                                 _Float16* __restrict__ y16) {
  const int D = D_MODEL;
  int row = blockIdx.x;
  const float* xr = x + (size_t)row * D;
  float s = 0.f, s2 = 0.f;
  for (int i = threadIdx.x; i < D; i += 256) {
    float v = xr[i];
    s += v;
    s2 += v * v;
  }
  __shared__ float rs[256], rq[256];
  rs[threadIdx.x] = s;
  rq[threadIdx.x] = s2;
  __syncthreads();
  for (int off = 128; off > 0; off >>= 1) {
    if (threadIdx.x < off) {
      rs[threadIdx.x] += rs[threadIdx.x + off];
      rq[threadIdx.x] += rq[threadIdx.x + off];
    }
    __syncthreads();
  }
  float mean = rs[0] / D;
  float var  = rq[0] / D - mean * mean;
  float inv  = rsqrtf(var + 1e-5f);
  for (int i = threadIdx.x; i < D; i += 256) {
    float v = (xr[i] - mean) * inv * g[i] + b[i];
    if (y32) y32[(size_t)row * D + i] = v;
    if (y16) y16[(size_t)row * D + i] = (_Float16)v;
  }
}

// ---------------------------------------------------------------- Hexagram projection
// One wave per token. Produces hex_embed (N,6) f32 and analogy weights aw (N,64) f16.
__global__ __launch_bounds__(256) void hex_kernel(const float* __restrict__ h,
                                                  const float* __restrict__ w_to6,
                                                  const float* __restrict__ hexa,
                                                  const float* __restrict__ log_temp,
                                                  float* __restrict__ he,
                                                  _Float16* __restrict__ aw16) {
  int lane = threadIdx.x & 31;
  int wave = threadIdx.x >> 5;
  int n = blockIdx.x * 8 + wave;
  const float* hr = h + (size_t)n * D_MODEL;

  float sb[6] = {0.f, 0.f, 0.f, 0.f, 0.f, 0.f};
  for (int k = lane; k < D_MODEL; k += 32) {
    float v = hr[k];
#pragma unroll
    for (int j = 0; j < 6; ++j) sb[j] += v * w_to6[j * D_MODEL + k];
  }
#pragma unroll
  for (int j = 0; j < 6; ++j) sb[j] = tanhf(redsum32(sb[j]));

  float tmp = __expf(*log_temp);
  tmp = fminf(fmaxf(tmp, 0.1f), 5.0f);
  float invT = 1.0f / tmp;

  int j0 = lane, j1 = lane + 32;
  float l0 = 0.f, l1 = 0.f;
#pragma unroll
  for (int c = 0; c < 6; ++c) {
    l0 += sb[c] * hexa[j0 * 6 + c];
    l1 += sb[c] * hexa[j1 * 6 + c];
  }
  l0 *= invT;
  l1 *= invT;
  float mx = redmax32(fmaxf(l0, l1));
  float e0 = __expf(l0 - mx), e1 = __expf(l1 - mx);
  float ssum = redsum32(e0 + e1);
  float w0 = e0 / ssum, w1 = e1 / ssum;

  // hex_embed = hex_w @ hexa  (== soft_hex in the reference)
#pragma unroll
  for (int c = 0; c < 6; ++c) {
    float p = redsum32(w0 * hexa[j0 * 6 + c] + w1 * hexa[j1 * 6 + c]);
    if (lane == 0) he[(size_t)n * 6 + c] = p;
  }

  // aw[l] = sum over hypercube neighbors (hamming distance 1), normalized
  float a0 = w1;  // neighbor l0 ^ 32 lives in this lane's w1
  float a1 = w0;  // neighbor l1 ^ 32 lives in this lane's w0
#pragma unroll
  for (int bb = 0; bb < 5; ++bb) {
    a0 += __shfl_xor(w0, 1 << bb, 32);
    a1 += __shfl_xor(w1, 1 << bb, 32);
  }
  float tot = redsum32(a0 + a1);
  float invA = 1.0f / (tot + 1e-8f);
  aw16[(size_t)n * 64 + j0] = (_Float16)(a0 * invA);
  aw16[(size_t)n * 64 + j1] = (_Float16)(a1 * invA);
}

// h_enr = h + sigmoid(gate) * (hex_embed @ w_from6^T)
__global__ __launch_bounds__(256) void hex_enrich_kernel(const float* __restrict__ h,
                                                         const float* __restrict__ he,
                                                         const float* __restrict__ w_from6,
                                                         const float* __restrict__ gate,
                                                         float* __restrict__ out) {
  int idx = blockIdx.x * 256 + threadIdx.x;
  int n = idx >> 10, i = idx & 1023;
  float g = sigmf(*gate);
  float acc = 0.f;
#pragma unroll
  for (int j = 0; j < 6; ++j) acc += he[(size_t)n * 6 + j] * w_from6[i * 6 + j];
  out[idx] = h[idx] + g * acc;
}

// ---------------------------------------------------------------- WMMA GEMM (f16 in, f32 acc)
// Out(M,Nout) = epi( alpha * A(M,K) x W(Nout,K)^T + Res )
// epi: 0 = plain, 1 = sigmoid, 2 = alpha*acc + Res.  Out (f32) and Out16 (f16) optional.
//
// Block tile 128x128, K-step 32, 8 waves (4M x 2N), 8 WMMAs per wave per K-step.
// Double-buffered LDS; staging uses GLOBAL_LOAD_ASYNC_TO_LDS_B128 when available.
__global__ __launch_bounds__(256) void gemm_kernel(const _Float16* __restrict__ A,
                                                   const _Float16* __restrict__ W,
                                                   float* __restrict__ Out,
                                                   _Float16* __restrict__ Out16, int M,
                                                   int Nout, int K, int epi,
                                                   const float* __restrict__ Res,
                                                   const float* __restrict__ scale_ptr,
                                                   int scale_op) {
  __shared__ __align__(16) _Float16 lA[2][128 * 40];
  __shared__ __align__(16) _Float16 lB[2][128 * 40];

  int tid = threadIdx.x;
  int lane = tid & 31;
  int wave = tid >> 5;
  int wm = wave & 3;   // 32 rows per wave
  int wn = wave >> 2;  // 64 cols per wave
  int l15 = lane & 15, hi = (lane >> 4) & 1;

  int mb = blockIdx.y * 128;
  int nb = blockIdx.x * 128;

  // staging: thread handles chunks (row, part) and (row+64, part); 16B per chunk
  int rowS = tid >> 2;        // 0..63
  int part = (tid & 3) * 8;   // halves offset 0,8,16,24

  v8f acc[2][4];
#pragma unroll
  for (int i = 0; i < 2; ++i)
#pragma unroll
    for (int j = 0; j < 4; ++j)
#pragma unroll
      for (int r = 0; r < 8; ++r) acc[i][j][r] = 0.f;

  const int kTiles = K / 32;

#if USE_ASYNC_LDS
  auto issue_tile = [&](int buf, int k0) {
#pragma unroll
    for (int c = 0; c < 2; ++c) {
      int row = rowS + c * 64;
      __builtin_amdgcn_global_load_async_to_lds_b128(
          (gv4i*)(A + (size_t)(mb + row) * K + k0 + part),
          (lv4i*)(&lA[buf][row * 40 + part]), 0, 0);
      __builtin_amdgcn_global_load_async_to_lds_b128(
          (gv4i*)(W + (size_t)(nb + row) * K + k0 + part),
          (lv4i*)(&lB[buf][row * 40 + part]), 0, 0);
    }
  };

  issue_tile(0, 0);
  for (int kt = 0; kt < kTiles; ++kt) {
    int cur = kt & 1;
    bool havePre = (kt + 1) < kTiles;
    if (havePre) {
      issue_tile(cur ^ 1, (kt + 1) * 32);
      if (kt + 2 < kTiles) {
        __builtin_prefetch(A + (size_t)(mb + rowS) * K + (kt + 2) * 32 + part, 0, 1);
        __builtin_prefetch(W + (size_t)(nb + rowS) * K + (kt + 2) * 32 + part, 0, 1);
      }
    }
    if (havePre) __builtin_amdgcn_s_wait_asynccnt(4);  // this tile's copies done
    else         __builtin_amdgcn_s_wait_asynccnt(0);
    __syncthreads();                                   // everyone's copies visible

    v16h af[2], bf[4];
#pragma unroll
    for (int i = 0; i < 2; ++i) af[i] = frag_lds40(&lA[cur][(wm * 32 + i * 16) * 40], lane);
#pragma unroll
    for (int j = 0; j < 4; ++j) bf[j] = frag_lds40(&lB[cur][(wn * 64 + j * 16) * 40], lane);
#pragma unroll
    for (int i = 0; i < 2; ++i)
#pragma unroll
      for (int j = 0; j < 4; ++j) acc[i][j] = WMMA_F16(af[i], bf[j], acc[i][j]);

    __syncthreads();  // reads of `cur` done before kt+1 DMAs into it
  }
#else
  auto load_regs = [&](int k0, v8h aR[2], v8h bR[2]) {
#pragma unroll
    for (int c = 0; c < 2; ++c) {
      int row = rowS + c * 64;
      aR[c] = *(const v8h*)(A + (size_t)(mb + row) * K + k0 + part);
      bR[c] = *(const v8h*)(W + (size_t)(nb + row) * K + k0 + part);
    }
  };
  auto store_lds = [&](int buf, const v8h aR[2], const v8h bR[2]) {
#pragma unroll
    for (int c = 0; c < 2; ++c) {
      int row = rowS + c * 64;
      *(v8h*)(&lA[buf][row * 40 + part]) = aR[c];
      *(v8h*)(&lB[buf][row * 40 + part]) = bR[c];
    }
  };

  {
    v8h aR[2], bR[2];
    load_regs(0, aR, bR);
    store_lds(0, aR, bR);
  }
  __syncthreads();

  for (int kt = 0; kt < kTiles; ++kt) {
    int cur = kt & 1;
    v8h aR[2], bR[2];
    bool havePre = (kt + 1) < kTiles;
    if (havePre) {
      load_regs((kt + 1) * 32, aR, bR);
      if (kt + 2 < kTiles) {
        __builtin_prefetch(A + (size_t)(mb + rowS) * K + (kt + 2) * 32 + part, 0, 1);
        __builtin_prefetch(W + (size_t)(nb + rowS) * K + (kt + 2) * 32 + part, 0, 1);
      }
    }

    v16h af[2], bf[4];
#pragma unroll
    for (int i = 0; i < 2; ++i) af[i] = frag_lds40(&lA[cur][(wm * 32 + i * 16) * 40], lane);
#pragma unroll
    for (int j = 0; j < 4; ++j) bf[j] = frag_lds40(&lB[cur][(wn * 64 + j * 16) * 40], lane);
#pragma unroll
    for (int i = 0; i < 2; ++i)
#pragma unroll
      for (int j = 0; j < 4; ++j) acc[i][j] = WMMA_F16(af[i], bf[j], acc[i][j]);

    if (havePre) store_lds(cur ^ 1, aR, bR);
    __syncthreads();
  }
#endif

  float alpha = (scale_op == 1 && scale_ptr) ? sigmf(*scale_ptr) : 1.0f;
  int mBase = mb + wm * 32;
  int nBase = nb + wn * 64;
#pragma unroll
  for (int i = 0; i < 2; ++i)
#pragma unroll
    for (int j = 0; j < 4; ++j)
#pragma unroll
      for (int r = 0; r < 8; ++r) {
        int row = mBase + i * 16 + hi * 8 + r;
        int col = nBase + j * 16 + l15;
        size_t idx = (size_t)row * Nout + col;
        float v = acc[i][j][r];
        if (epi == 1) v = sigmf(v);
        else if (epi == 2) v = alpha * v + Res[idx];
        if (Out) Out[idx] = v;
        if (Out16) Out16[idx] = (_Float16)v;
      }
}

// ---------------------------------------------------------------- Flash attention
// One wave per (b, head, 16-query tile). Causal + hamming bias. Q/K/V f16, output f16.
__global__ __launch_bounds__(256) void attn_kernel(const _Float16* __restrict__ Q,
                                                   const _Float16* __restrict__ Kp,
                                                   const _Float16* __restrict__ V,
                                                   const float* __restrict__ he,
                                                   _Float16* __restrict__ O,
                                                   const float* __restrict__ ham_logit) {
  __shared__ __align__(16) _Float16 ldsP[8 * 16 * 40];

  const int D = D_MODEL, T = T_SEQ;
  int lane = threadIdx.x & 31;
  int wave = threadIdx.x >> 5;
  int tile = blockIdx.x * 8 + wave;
  int qt = tile & 63;
  int bh = tile >> 6;
  int h = bh & (N_HEADS - 1);
  int b = bh >> 4;
  int qbase = qt * 16;
  int l15 = lane & 15, hi = (lane >> 4) & 1;

  _Float16* myP = &ldsP[wave * 640];

  const _Float16* qrow = Q + (size_t)(b * T + qbase) * D + h * 64;
  v16h q0 = frag_g16(qrow + 0, D, lane);
  v16h q1 = frag_g16(qrow + 32, D, lane);

  float heq[8][6];
#pragma unroll
  for (int r = 0; r < 8; ++r) {
    int t = qbase + hi * 8 + r;
#pragma unroll
    for (int c = 0; c < 6; ++c) heq[r][c] = he[(size_t)(b * T + t) * 6 + c];
  }
  float hamc = sigmf(*ham_logit) * 0.5f;

  v8f o[4];
  float m_old[8], l_run[8];
#pragma unroll
  for (int j = 0; j < 4; ++j)
#pragma unroll
    for (int r = 0; r < 8; ++r) o[j][r] = 0.f;
#pragma unroll
  for (int r = 0; r < 8; ++r) { m_old[r] = -3.0e38f; l_run[r] = 0.f; }

  for (int s0 = 0; s0 <= qbase + 15; s0 += 32) {
    const _Float16* krow0 = Kp + (size_t)(b * T + s0) * D + h * 64;
    const _Float16* krow1 = Kp + (size_t)(b * T + s0 + 16) * D + h * 64;
    v16h ka0 = frag_g16(krow0 + 0, D, lane);
    v16h ka1 = frag_g16(krow0 + 32, D, lane);
    v16h kb0 = frag_g16(krow1 + 0, D, lane);
    v16h kb1 = frag_g16(krow1 + 32, D, lane);

    v8f S0, S1;
#pragma unroll
    for (int r = 0; r < 8; ++r) { S0[r] = 0.f; S1[r] = 0.f; }
    S0 = WMMA_F16(q0, ka0, S0);
    S0 = WMMA_F16(q1, ka1, S0);
    S1 = WMMA_F16(q0, kb0, S1);
    S1 = WMMA_F16(q1, kb1, S1);

    float hes0[6], hes1[6];
#pragma unroll
    for (int c = 0; c < 6; ++c) {
      hes0[c] = he[(size_t)(b * T + s0 + l15) * 6 + c];
      hes1[c] = he[(size_t)(b * T + s0 + 16 + l15) * 6 + c];
    }

    int sA = s0 + l15, sB = s0 + 16 + l15;
#pragma unroll
    for (int r = 0; r < 8; ++r) {
      int t = qbase + hi * 8 + r;
      float d0 = 0.f, d1 = 0.f;
#pragma unroll
      for (int c = 0; c < 6; ++c) {
        d0 += heq[r][c] * hes0[c];
        d1 += heq[r][c] * hes1[c];
      }
      float v0 = S0[r] * 0.125f + hamc * d0;
      float v1 = S1[r] * 0.125f + hamc * d1;
      if (sA > t) v0 = -3.0e38f;
      if (sB > t) v1 = -3.0e38f;
      S0[r] = v0;
      S1[r] = v1;
    }

#pragma unroll
    for (int r = 0; r < 8; ++r) {
      float rm = redmax16(fmaxf(S0[r], S1[r]));
      float mn = fmaxf(m_old[r], rm);
      float cf = __expf(m_old[r] - mn);
      float e0 = __expf(S0[r] - mn);
      float e1 = __expf(S1[r] - mn);
      float rsum = redsum16(e0 + e1);
      l_run[r] = l_run[r] * cf + rsum;
      m_old[r] = mn;
#pragma unroll
      for (int j = 0; j < 4; ++j) o[j][r] *= cf;
      myP[(hi * 8 + r) * 40 + l15]      = (_Float16)e0;
      myP[(hi * 8 + r) * 40 + 16 + l15] = (_Float16)e1;
    }

    v16h pf = frag_lds40(myP, lane);  // P: 16 x 32
    const _Float16* vb = V + (size_t)(b * T + s0) * D + h * 64;
    o[0] = WMMA_F16(pf, frag_vcol16(vb + 0, D, lane), o[0]);
    o[1] = WMMA_F16(pf, frag_vcol16(vb + 16, D, lane), o[1]);
    o[2] = WMMA_F16(pf, frag_vcol16(vb + 32, D, lane), o[2]);
    o[3] = WMMA_F16(pf, frag_vcol16(vb + 48, D, lane), o[3]);
  }

#pragma unroll
  for (int r = 0; r < 8; ++r) {
    float inv = l_run[r] > 0.f ? 1.0f / l_run[r] : 0.f;
    int row = b * T + qbase + hi * 8 + r;
#pragma unroll
    for (int j = 0; j < 4; ++j)
      O[(size_t)row * D + h * 64 + j * 16 + l15] = (_Float16)(o[j][r] * inv);
  }
}

// ---------------------------------------------------------------- elementwise
__global__ __launch_bounds__(256) void tern_kernel(const float* __restrict__ x1,
                                                   const float* __restrict__ s,
                                                   const float* __restrict__ log_temp,
                                                   const float* __restrict__ log_unc,
                                                   const float* __restrict__ scale,
                                                   float* __restrict__ out,
                                                   _Float16* __restrict__ out16) {
  int idx = blockIdx.x * 256 + threadIdx.x;
  float tt = fminf(fmaxf(__expf(*log_temp), 0.05f), 2.0f);
  float ub = sigmf(*log_unc);
  float thr = (1.0f - ub) * 0.5f + 0.1f;
  float coef = (*scale) * (1.0f - ub * 0.5f);
  float sc = tanhf(s[idx] / tt);
  float hard = sc > thr ? 1.0f : (sc < -thr ? -1.0f : 0.0f);
  float v = x1[idx] * (1.0f + coef * hard);
  out[idx] = v;
  out16[idx] = (_Float16)v;
}

__global__ __launch_bounds__(256) void mul16_kernel(const float* __restrict__ a,
                                                    const float* __restrict__ b,
                                                    _Float16* __restrict__ out16) {
  int idx = blockIdx.x * 256 + threadIdx.x;
  out16[idx] = (_Float16)(a[idx] * b[idx]);
}

__global__ __launch_bounds__(256) void silu_mul16_kernel(const float* __restrict__ g,
                                                         const float* __restrict__ v,
                                                         _Float16* __restrict__ out16) {
  int idx = blockIdx.x * 256 + threadIdx.x;
  float x = g[idx];
  out16[idx] = (_Float16)((x * sigmf(x)) * v[idx]);
}

// ---------------------------------------------------------------- host driver
static inline void launch_gemm(hipStream_t st, const _Float16* A, const _Float16* W, float* O,
                               _Float16* O16, int M, int Nout, int K, int epi, const float* Res,
                               const float* scale_ptr, int scale_op) {
  dim3 grid(Nout / 128, M / 128);
  gemm_kernel<<<grid, dim3(256), 0, st>>>(A, W, O, O16, M, Nout, K, epi, Res, scale_ptr,
                                          scale_op);
}

extern "C" void kernel_launch(void* const* d_in, const int* in_sizes, int n_in, void* d_out,
                              int out_size, void* d_ws, size_t ws_size, hipStream_t stream) {
  (void)in_sizes; (void)n_in; (void)out_size; (void)ws_size;

  const float* x           = (const float*)d_in[0];
  const float* ln_hex_g    = (const float*)d_in[1];
  const float* ln_hex_b    = (const float*)d_in[2];
  const float* ln_attn_g   = (const float*)d_in[3];
  const float* ln_attn_b   = (const float*)d_in[4];
  const float* ln_ffn_g    = (const float*)d_in[5];
  const float* ln_ffn_b    = (const float*)d_in[6];
  const float* hex_w_to6   = (const float*)d_in[7];
  const float* hex_w_from6 = (const float*)d_in[8];
  const float* hex_gate    = (const float*)d_in[9];
  const float* hex_log_t   = (const float*)d_in[10];
  const float* wq          = (const float*)d_in[11];
  const float* wk          = (const float*)d_in[12];
  const float* wv          = (const float*)d_in[13];
  const float* wo          = (const float*)d_in[14];
  const float* ham_logit   = (const float*)d_in[15];
  const float* tg_w        = (const float*)d_in[16];
  const float* tg_log_unc  = (const float*)d_in[17];
  const float* tg_log_temp = (const float*)d_in[18];
  const float* tg_scale    = (const float*)d_in[19];
  const float* an_embed    = (const float*)d_in[20];
  const float* an_wg       = (const float*)d_in[21];
  const float* an_wo       = (const float*)d_in[22];
  const float* an_scale    = (const float*)d_in[23];
  const float* ffn_wg      = (const float*)d_in[24];
  const float* ffn_wv      = (const float*)d_in[25];
  const float* ffn_wo      = (const float*)d_in[26];
  const float* hexa        = (const float*)d_in[27];
  // d_in[28] = biangua: unused (hamming-1 adjacency == hypercube neighbors via shfl_xor)

  float* out = (float*)d_out;

  // ---- workspace carve-up
  const size_t NB = (size_t)N_TOK * D_MODEL;  // 4M elems
  const size_t NF = (size_t)N_TOK * D_FF;     // 16M elems
  const size_t NW = (size_t)D_MODEL * D_MODEL;
  const size_t NWF = (size_t)D_FF * D_MODEL;
  float* ws = (float*)d_ws;
  // f32 region
  float* B0 = ws + 0 * NB;  // h -> tg_s -> (scratch)
  float* B1 = ws + 1 * NB;  // h_enr -> ae
  float* B2 = ws + 2 * NB;  // agate
  float* B3 = ws + 3 * NB;  // x2
  float* B6 = ws + 4 * NB;  // x3
  float* B7 = ws + 5 * NB;  // x1
  float* F0 = ws + 6 * NB;            // (N, DFF)
  float* F1 = F0 + NF;
  float* S_he = F1 + NF;              // (N,6) in N*8 region
  // f16 region
  _Float16* hp = (_Float16*)(S_he + (size_t)N_TOK * 8);
  _Float16* wq16  = hp; hp += NW;
  _Float16* wk16  = hp; hp += NW;
  _Float16* wv16  = hp; hp += NW;
  _Float16* wo16  = hp; hp += NW;
  _Float16* tg16  = hp; hp += NW;
  _Float16* ang16 = hp; hp += NW;
  _Float16* ano16 = hp; hp += NW;
  _Float16* wg16  = hp; hp += NWF;
  _Float16* wvf16 = hp; hp += NWF;
  _Float16* wof16 = hp; hp += NWF;
  _Float16* anET  = hp; hp += (size_t)D_MODEL * 64;
  _Float16* a_in16 = hp; hp += NB;
  _Float16* Q16    = hp; hp += NB;
  _Float16* K16    = hp; hp += NB;
  _Float16* V16    = hp; hp += NB;
  _Float16* aO16   = hp; hp += NB;
  _Float16* x1_16  = hp; hp += NB;
  _Float16* x2_16  = hp; hp += NB;
  _Float16* gae16  = hp; hp += NB;
  _Float16* f16b   = hp; hp += NB;
  _Float16* gv16   = hp; hp += NF;
  _Float16* aw16   = hp; hp += (size_t)N_TOK * 64;

  const int ND_BLK  = (N_TOK * D_MODEL) / 256;
  const int NFF_BLK = (N_TOK * D_FF) / 256;

  // ---- weight conversion (f32 -> f16), once per call
  cvt16_kernel<<<NW / 1024, 256, 0, stream>>>(wq, wq16);
  cvt16_kernel<<<NW / 1024, 256, 0, stream>>>(wk, wk16);
  cvt16_kernel<<<NW / 1024, 256, 0, stream>>>(wv, wv16);
  cvt16_kernel<<<NW / 1024, 256, 0, stream>>>(wo, wo16);
  cvt16_kernel<<<NW / 1024, 256, 0, stream>>>(tg_w, tg16);
  cvt16_kernel<<<NW / 1024, 256, 0, stream>>>(an_wg, ang16);
  cvt16_kernel<<<NW / 1024, 256, 0, stream>>>(an_wo, ano16);
  cvt16_kernel<<<NWF / 1024, 256, 0, stream>>>(ffn_wg, wg16);
  cvt16_kernel<<<NWF / 1024, 256, 0, stream>>>(ffn_wv, wvf16);
  cvt16_kernel<<<NWF / 1024, 256, 0, stream>>>(ffn_wo, wof16);
  tconv_kernel<<<(D_MODEL * 64) / 256, 256, 0, stream>>>(an_embed, anET);

  // ---- pipeline
  // 1. h = LN_hex(x)
  ln_kernel<<<N_TOK, 256, 0, stream>>>(x, ln_hex_g, ln_hex_b, B0, nullptr);
  // 2. hexagram projection
  hex_kernel<<<N_TOK / 8, 256, 0, stream>>>(B0, hex_w_to6, hexa, hex_log_t, S_he, aw16);
  // 3. h_enr = h + sig(gate) * he @ w_from6^T
  hex_enrich_kernel<<<ND_BLK, 256, 0, stream>>>(B0, S_he, hex_w_from6, hex_gate, B1);
  // 4. a_in = LN_attn(h_enr)  (f16 only)
  ln_kernel<<<N_TOK, 256, 0, stream>>>(B1, ln_attn_g, ln_attn_b, nullptr, a_in16);
  // 5. q, k, v  (f16 only)
  launch_gemm(stream, a_in16, wq16, nullptr, Q16, N_TOK, D_MODEL, D_MODEL, 0, nullptr, nullptr, 0);
  launch_gemm(stream, a_in16, wk16, nullptr, K16, N_TOK, D_MODEL, D_MODEL, 0, nullptr, nullptr, 0);
  launch_gemm(stream, a_in16, wv16, nullptr, V16, N_TOK, D_MODEL, D_MODEL, 0, nullptr, nullptr, 0);
  // 6. causal attention with hamming bias
  attn_kernel<<<(NBATCH * N_HEADS * (T_SEQ / 16)) / 8, 256, 0, stream>>>(Q16, K16, V16, S_he,
                                                                         aO16, ham_logit);
  // 7. x1 = x + attn_out @ wo^T  (f32 + f16)
  launch_gemm(stream, aO16, wo16, B7, x1_16, N_TOK, D_MODEL, D_MODEL, 2, x, nullptr, 0);
  // 8. ternary gate
  launch_gemm(stream, x1_16, tg16, B0, nullptr, N_TOK, D_MODEL, D_MODEL, 0, nullptr, nullptr, 0);
  tern_kernel<<<ND_BLK, 256, 0, stream>>>(B7, B0, tg_log_temp, tg_log_unc, tg_scale, B3, x2_16);
  // 9. analogy: ae = aw @ an_embed
  launch_gemm(stream, aw16, anET, B1, nullptr, N_TOK, D_MODEL, 64, 0, nullptr, nullptr, 0);
  // 10. agate = sigmoid(x2 @ an_wg^T); gae = agate * ae
  launch_gemm(stream, x2_16, ang16, B2, nullptr, N_TOK, D_MODEL, D_MODEL, 1, nullptr, nullptr, 0);
  mul16_kernel<<<ND_BLK, 256, 0, stream>>>(B2, B1, gae16);
  // 11. x3 = x2 + sig(an_scale) * (gae @ an_wo^T)
  launch_gemm(stream, gae16, ano16, B6, nullptr, N_TOK, D_MODEL, D_MODEL, 2, B3, an_scale, 1);
  // 12. f = LN_ffn(x3)  (f16 only)
  ln_kernel<<<N_TOK, 256, 0, stream>>>(B6, ln_ffn_g, ln_ffn_b, nullptr, f16b);
  // 13. SwiGLU FFN
  launch_gemm(stream, f16b, wg16, F0, nullptr, N_TOK, D_FF, D_MODEL, 0, nullptr, nullptr, 0);
  launch_gemm(stream, f16b, wvf16, F1, nullptr, N_TOK, D_FF, D_MODEL, 0, nullptr, nullptr, 0);
  silu_mul16_kernel<<<NFF_BLK, 256, 0, stream>>>(F0, F1, gv16);
  // 14. out = x3 + gv @ ffn_wo^T
  launch_gemm(stream, gv16, wof16, out, nullptr, N_TOK, D_MODEL, D_FF, 2, B6, nullptr, 0);
}
